// Glm4MoeSparseMoeBlock_64939905516054
// MI455X (gfx1250) — matrix operand order repaired and checked by reading, not compiled, not run on hardware
//
#include <hip/hip_runtime.h>
#include <hip/hip_bf16.h>
#include <cstdint>

// ---------------- problem constants ----------------
constexpr int T = 1024;
constexpr int H = 1024;
constexpr int E = 64;
constexpr int I = 512;
constexpr int G = 8;
constexpr int TOPKG = 4;
constexpr int K = 8;
constexpr int ROWS_CAP = 12288;   // >= sum(ceil16(Ne)) + 1024 shared rows (max 10176)

typedef __attribute__((ext_vector_type(16))) __bf16 v16bf;
typedef __attribute__((ext_vector_type(8)))  __bf16 v8bf;
typedef __attribute__((ext_vector_type(8)))  float  v8f;

__device__ inline v8f wmma_bf16(v16bf a, v16bf b, v8f c) {
  return __builtin_amdgcn_wmma_f32_16x16x32_bf16(false, a, false, b, (short)0, c,
                                                 false, false);
}

__device__ inline float fast_rcp(float x) { return __builtin_amdgcn_rcpf(x); }

union AFrag { v16bf v; v8bf h[2]; };

// =====================================================================
// 1) Gating: logits -> sigmoid -> grouped top-k routing
// =====================================================================
__global__ __launch_bounds__(64) void k_gate(const float* __restrict__ x,
                                             const float* __restrict__ gw,
                                             const float* __restrict__ eb,
                                             int* __restrict__ tki,
                                             float* __restrict__ tkw) {
  const int t = blockIdx.x;
  const int e = threadIdx.x;           // 64 threads, one expert each
  __shared__ float sx[H];              // staged token row
  __shared__ float ls[E];              // sigmoid scores
  __shared__ float lc[E];              // score + bias (selection), later masked

  const float* xr = x + (size_t)t * H;
  // stage the token row once (float4 coalesced), all 64 threads dot from LDS
  for (int i = e; i < H / 4; i += 64)
    ((float4*)sx)[i] = ((const float4*)xr)[i];
  __syncthreads();

  const float* wr = gw + (size_t)e * H;
  float acc = 0.f;
  #pragma unroll 4
  for (int k = 0; k < H; k += 4) {
    acc += sx[k] * wr[k] + sx[k + 1] * wr[k + 1] + sx[k + 2] * wr[k + 2] +
           sx[k + 3] * wr[k + 3];
  }
  const float s = fast_rcp(1.f + __expf(-acc));
  ls[e] = s;
  lc[e] = s + eb[e];
  __syncthreads();

  if (e == 0) {
    // group scores: sum of top-2 per group of 8
    float gs[G];
    for (int g = 0; g < G; ++g) {
      float m1 = -1e30f, m2 = -1e30f;
      for (int j = 0; j < 8; ++j) {
        float v = lc[g * 8 + j];
        if (v > m1) { m2 = m1; m1 = v; }
        else if (v > m2) { m2 = v; }
      }
      gs[g] = m1 + m2;
    }
    // top-4 groups (strict > keeps lowest index on ties, like lax.top_k)
    bool sel[G];
    for (int g = 0; g < G; ++g) sel[g] = false;
    for (int it = 0; it < TOPKG; ++it) {
      float best = -3e30f; int bi = 0;
      for (int g = 0; g < G; ++g)
        if (!sel[g] && gs[g] > best) { best = gs[g]; bi = g; }
      sel[bi] = true;
    }
    // mask non-selected groups
    for (int i = 0; i < E; ++i)
      if (!sel[i >> 3]) lc[i] = -1e30f;
    // top-8 experts over masked selection scores
    int idx[K];
    for (int it = 0; it < K; ++it) {
      float best = -3e30f; int bi = 0;
      for (int i = 0; i < E; ++i)
        if (lc[i] > best) { best = lc[i]; bi = i; }
      idx[it] = bi;
      lc[bi] = -3e30f;
    }
    float w[K], wsum = 0.f;
    for (int k = 0; k < K; ++k) { w[k] = ls[idx[k]]; wsum += w[k]; }
    const float inv = fast_rcp(wsum);   // RSF = 1.0
    for (int k = 0; k < K; ++k) {
      tki[t * K + k] = idx[k];
      tkw[t * K + k] = w[k] * inv;
    }
  }
}

// =====================================================================
// 2) Per-expert token counts (deterministic)
// =====================================================================
__global__ __launch_bounds__(1024) void k_count(const int* __restrict__ tki,
                                                int* __restrict__ counts) {
  const int e = blockIdx.x;        // 0..63
  const int t = threadIdx.x;       // one token per thread
  int match = 0;
  #pragma unroll
  for (int k = 0; k < K; ++k)
    if (tki[t * K + k] == e) match = 1;
  unsigned long long b = __ballot(match);
  __shared__ int wsum[32];
  const int lane = t & 31, wid = t >> 5;
  if (lane == 0) wsum[wid] = (int)__popcll(b);
  __syncthreads();
  if (t == 0) {
    int tot = 0;
    for (int i = 0; i < 32; ++i) tot += wsum[i];
    counts[e] = tot;
  }
}

__global__ void k_prefix(const int* __restrict__ counts, int* __restrict__ basep) {
  if (threadIdx.x == 0 && blockIdx.x == 0) {
    int run = 0;
    for (int e = 0; e < E; ++e) {
      basep[e] = run;
      run += (counts[e] + 15) & ~15;
    }
    basep[E] = run;                  // shared-expert row block (1024 rows)
  }
}

// =====================================================================
// 3) Fill compacted row lists (sorted by token id -> deterministic)
// =====================================================================
__global__ __launch_bounds__(1024) void k_fill(const int* __restrict__ tki,
                                               const float* __restrict__ tkw,
                                               const int* __restrict__ counts,
                                               const int* __restrict__ basep,
                                               int* __restrict__ tok,
                                               float* __restrict__ scl,
                                               int* __restrict__ posof) {
  const int e = blockIdx.x;        // 0..64 (64 == shared expert)
  const int t = threadIdx.x;
  if (e == E) {
    const int b64 = basep[E];
    tok[b64 + t] = t;
    scl[b64 + t] = 1.f;
    return;
  }
  int match = 0, mk = 0;
  float w = 0.f;
  #pragma unroll
  for (int k = 0; k < K; ++k) {
    if (tki[t * K + k] == e) { match = 1; mk = k; w = tkw[t * K + k]; }
  }
  unsigned long long b = __ballot(match);
  __shared__ int wsum[32];
  __shared__ int woff[32];
  const int lane = t & 31, wid = t >> 5;
  if (lane == 0) wsum[wid] = (int)__popcll(b);
  __syncthreads();
  if (t == 0) {
    int run = 0;
    for (int i = 0; i < 32; ++i) { woff[i] = run; run += wsum[i]; }
  }
  __syncthreads();
  const int base = basep[e];
  const int pos =
      base + woff[wid] + (int)__popcll(b & ((1ull << lane) - 1ull));
  if (match) {
    tok[pos] = t;
    scl[pos] = w;
    posof[t * K + mk] = pos;
  }
  // pad rows up to multiple of 16
  const int cnt = counts[e];
  const int padn = ((cnt + 15) & ~15) - cnt;
  if (t < padn) {
    tok[base + cnt + t] = -1;
    scl[base + cnt + t] = 0.f;
  }
}

// =====================================================================
// 4) GEMM1: act = silu(X W_g) * (X W_u) * route_weight   (bf16 WMMA)
//    block = 8 waves; covers 64 rows x (128 gate + 128 up) cols.
//    4x M-subtiles per wave amortize the strided weight stream.
// =====================================================================
__global__ __launch_bounds__(256) void k_gemm1(
    const float* __restrict__ x, const float* __restrict__ wgu,
    const float* __restrict__ shgu, const int* __restrict__ counts,
    const int* __restrict__ basep, const int* __restrict__ tok,
    const float* __restrict__ scl, __bf16* __restrict__ act) {
  const int nt = blockIdx.x;   // 0..3  -> gate cols [nt*128, nt*128+128)
  const int e  = blockIdx.y;   // 0..64
  const int rt = blockIdx.z;   // 0..15 -> rows [rt*64, rt*64+64)
  const int npad = (e == E) ? T : ((counts[e] + 15) & ~15);
  const int r0 = rt * 64;
  if (r0 >= npad) return;
  const int nloc = npad - r0;                        // valid rows in chunk
  const int nlim = nloc < 64 ? nloc : 64;            // multiple of 16
  const int rowbase = basep[e] + r0;

  __shared__ __align__(16) __bf16 As[64][1032];      // padded stride
  __shared__ float sScale[64];
  __shared__ int   sTok[64];

  const int tid = threadIdx.x;
  if (tid < 64) {
    const int valid = tid < nlim;
    sTok[tid]   = valid ? tok[rowbase + tid] : -1;
    sScale[tid] = valid ? scl[rowbase + tid] : 0.f;
  }
  __syncthreads();

  // stage A: gather 64 token rows, fp32 -> bf16 into LDS (32768 pairs)
  #pragma unroll 4
  for (int i = 0; i < 128; ++i) {
    const int p   = tid + i * 256;
    const int row = p >> 9;            // 512 pairs per row
    const int col = (p & 511) * 2;
    float2 v = make_float2(0.f, 0.f);
    const int tk = sTok[row];
    if (tk >= 0) v = *(const float2*)(x + (size_t)tk * H + col);
    const unsigned short u0 = __builtin_bit_cast(unsigned short, (__bf16)v.x);
    const unsigned short u1 = __builtin_bit_cast(unsigned short, (__bf16)v.y);
    *(unsigned int*)&As[row][col] = (unsigned)u0 | ((unsigned)u1 << 16);
  }
  __syncthreads();

  const int lane = tid & 31;
  const int wv   = tid >> 5;
  const int colg = nt * 128 + wv * 16;               // gate col base, [0,512)
  const float* Wb = (e == E) ? shgu : (wgu + (size_t)e * H * (2 * I));

  const int am  = lane & 15;            // A: row within subtile
  const int akh = lane >> 4;            // A: K-half selector
  const int bn  = lane & 15;            // B: column
  const int bkh = lane >> 4;            // B: K-half selector

  v8f accg[4] = {};
  v8f accu[4] = {};
  for (int k0 = 0; k0 < H; k0 += 32) {
    v16bf bg, bu;
    const float* wp  = Wb + (size_t)(k0 + bkh * 16) * (2 * I) + colg + bn;
    const float* wpu = wp + I;
    __builtin_prefetch(wp + (size_t)32 * (2 * I), 0, 0);   // next K-step
    __builtin_prefetch(wpu + (size_t)32 * (2 * I), 0, 0);
    #pragma unroll
    for (int j = 0; j < 16; ++j) bg[j] = (__bf16)wp[(size_t)j * (2 * I)];
    #pragma unroll
    for (int j = 0; j < 16; ++j) bu[j] = (__bf16)wpu[(size_t)j * (2 * I)];
    #pragma unroll
    for (int s = 0; s < 4; ++s) {
      AFrag a;
      a.h[0] = *(const v8bf*)&As[s * 16 + am][k0 + akh * 16];
      a.h[1] = *(const v8bf*)&As[s * 16 + am][k0 + akh * 16 + 8];
      accg[s] = wmma_bf16(a.v, bg, accg[s]);
      accu[s] = wmma_bf16(a.v, bu, accu[s]);
    }
  }

  // epilogue: silu(g) * u * route_weight -> bf16 act
  // nlim is a multiple of 16, so the row guard is uniform per 16-row subtile
  const int N  = lane & 15;
  const int mh = (lane >> 4) * 8;
  #pragma unroll
  for (int s = 0; s < 4; ++s) {
    if (s * 16 < nlim) {
      #pragma unroll
      for (int r = 0; r < 8; ++r) {
        const int M = s * 16 + r + mh;
        const float g = accg[s][r];
        const float u = accu[s][r];
        const float a = g * fast_rcp(1.f + __expf(-g)) * u * sScale[M];
        act[(size_t)(rowbase + M) * I + colg + N] = (__bf16)a;
      }
    }
  }
}

// =====================================================================
// 5) GEMM2: down = act @ W_down  (bf16 WMMA, f32 out to workspace)
//    block = 8 waves; 64 rows x 128 H-cols, 4x M-subtiles per wave.
// =====================================================================
__global__ __launch_bounds__(256) void k_gemm2(
    const __bf16* __restrict__ act, const float* __restrict__ wd,
    const float* __restrict__ shd, const int* __restrict__ counts,
    const int* __restrict__ basep, float* __restrict__ down) {
  const int nt = blockIdx.x;   // 0..7 -> H cols [nt*128, nt*128+128)
  const int e  = blockIdx.y;
  const int rt = blockIdx.z;   // 0..15 -> rows [rt*64, +64)
  const int npad = (e == E) ? T : ((counts[e] + 15) & ~15);
  const int r0 = rt * 64;
  if (r0 >= npad) return;
  const int nloc = npad - r0;
  const int nlim = nloc < 64 ? nloc : 64;            // multiple of 16
  const int rowbase = basep[e] + r0;

  const int tid  = threadIdx.x;
  const int lane = tid & 31;
  const int wv   = tid >> 5;
  const int colh = nt * 128 + wv * 16;
  const float* Wb = (e == E) ? shd : (wd + (size_t)e * I * H);

  const int am  = lane & 15;
  const int akh = lane >> 4;
  const int bn  = lane & 15;
  const int bkh = lane >> 4;

  v8f acc[4] = {};
  for (int k0 = 0; k0 < I; k0 += 32) {
    v16bf b;
    const float* wp = Wb + (size_t)(k0 + bkh * 16) * H + colh + bn;
    __builtin_prefetch(wp + (size_t)32 * H, 0, 0);         // next K-step
    #pragma unroll
    for (int j = 0; j < 16; ++j) b[j] = (__bf16)wp[(size_t)j * H];
    #pragma unroll
    for (int s = 0; s < 4; ++s) {
      const __bf16* ar = act + (size_t)(rowbase + s * 16 + am) * I;
      AFrag a;
      a.h[0] = *(const v8bf*)&ar[k0 + akh * 16];
      a.h[1] = *(const v8bf*)&ar[k0 + akh * 16 + 8];
      acc[s] = wmma_bf16(a.v, b, acc[s]);
    }
  }

  const int N  = lane & 15;
  const int mh = (lane >> 4) * 8;
  #pragma unroll
  for (int s = 0; s < 4; ++s) {
    if (s * 16 < nlim) {                               // uniform subtile guard
      #pragma unroll
      for (int r = 0; r < 8; ++r) {
        const int M = s * 16 + r + mh;
        down[(size_t)(rowbase + M) * H + colh + N] = acc[s][r];
      }
    }
  }
}

// =====================================================================
// 6) Combine: out[t] = shared_row(t) + sum_k down[pos(t,k)]  (deterministic)
// =====================================================================
__global__ __launch_bounds__(256) void k_combine(const float* __restrict__ down,
                                                 const int* __restrict__ posof,
                                                 const int* __restrict__ basep,
                                                 float* __restrict__ out) {
  const int t   = blockIdx.x;
  const int tid = threadIdx.x;          // 256 threads x float4 = H floats
  const float4* sh = (const float4*)(down + (size_t)(basep[E] + t) * H);
  float4 a = sh[tid];
  #pragma unroll
  for (int k = 0; k < K; ++k) {
    const int r = posof[t * K + k];
    const float4 v = ((const float4*)(down + (size_t)r * H))[tid];
    a.x += v.x; a.y += v.y; a.z += v.z; a.w += v.w;
  }
  ((float4*)(out + (size_t)t * H))[tid] = a;
}

// =====================================================================
// launch
// =====================================================================
extern "C" void kernel_launch(void* const* d_in, const int* in_sizes, int n_in,
                              void* d_out, int out_size, void* d_ws, size_t ws_size,
                              hipStream_t stream) {
  (void)in_sizes; (void)n_in; (void)out_size; (void)ws_size;
  const float* x    = (const float*)d_in[0];   // [T,H]
  const float* gw   = (const float*)d_in[1];   // [E,H]
  const float* eb   = (const float*)d_in[2];   // [E]
  const float* wgu  = (const float*)d_in[3];   // [E,H,2I]
  const float* wd   = (const float*)d_in[4];   // [E,I,H]
  const float* shgu = (const float*)d_in[5];   // [H,2I]
  const float* shd  = (const float*)d_in[6];   // [I,H]
  float* out = (float*)d_out;                  // [T,H]

  // ---- workspace carve-up (all 16B aligned) ----
  int*   tki    = (int*)d_ws;                    // T*K
  float* tkw    = (float*)(tki + T * K);         // T*K
  int*   counts = (int*)(tkw + T * K);           // 128
  int*   basep  = counts + 128;                  // 128
  int*   posof  = basep + 128;                   // T*K
  int*   tok    = posof + T * K;                 // ROWS_CAP
  float* scl    = (float*)(tok + ROWS_CAP);      // ROWS_CAP
  __bf16* act   = (__bf16*)(scl + ROWS_CAP);     // ROWS_CAP * I
  float* down   = (float*)(act + (size_t)ROWS_CAP * I);  // ROWS_CAP * H

  k_gate<<<T, 64, 0, stream>>>(x, gw, eb, tki, tkw);
  k_count<<<E, 1024, 0, stream>>>(tki, counts);
  k_prefix<<<1, 1, 0, stream>>>(counts, basep);
  k_fill<<<E + 1, 1024, 0, stream>>>(tki, tkw, counts, basep, tok, scl, posof);
  k_gemm1<<<dim3(4, E + 1, T / 64), 256, 0, stream>>>(x, wgu, shgu, counts,
                                                      basep, tok, scl, act);
  k_gemm2<<<dim3(8, E + 1, T / 64), 256, 0, stream>>>(act, wd, shd, counts,
                                                      basep, down);
  k_combine<<<T, 256, 0, stream>>>(down, posof, basep, out);
}
